// NaiveStackedLSTM_83056077570357
// MI455X (gfx1250) — matrix operand order repaired and checked by reading, not compiled, not run on hardware
//
#include <hip/hip_runtime.h>

typedef __attribute__((ext_vector_type(16))) __bf16 v16bf;
typedef __attribute__((ext_vector_type(8)))  float  v8f;

union Frag {
    uint4 q[2];
    v16bf v;
};

__device__ __forceinline__ unsigned short f2bf(float f) {
    unsigned int u = __float_as_uint(f);
    unsigned int r = u + 0x7FFFu + ((u >> 16) & 1u);   // round-to-nearest-even
    return (unsigned short)(r >> 16);
}

__device__ __forceinline__ float sigmoidf_fast(float x) {
    return 1.0f / (1.0f + __expf(-x));
}
__device__ __forceinline__ float tanhf_fast(float x) {
    // stable for large |x|: -> +/-1
    return 1.0f - 2.0f / (1.0f + __expf(2.0f * x));
}

// ---------------------------------------------------------------------------
// Pack [Wx;Wh] (f32, row-major [Ktot,4096]) into bf16 WMMA B-fragment order.
// For column-tile nt (16 cols), k-tile kt (32 rows):
//   dst[ (nt*Kt + kt)*512 + lane*16 + j ] = W[ kt*32 + (lane>>4)*8 + (j&7) + (j>>3)*16 ][ nt*16 + (lane&15) ]
// so each lane's 16-element fragment is 32 contiguous bytes.
// ---------------------------------------------------------------------------
__global__ void pack_weights_kernel(const float* __restrict__ Wx,
                                    const float* __restrict__ Wh,
                                    unsigned short* __restrict__ dst,
                                    int Kx, int Ktot, int total) {
    int idx = blockIdx.x * blockDim.x + threadIdx.x;
    if (idx >= total) return;
    const int Kt   = Ktot >> 5;          // k-tiles of 32
    const int perN = Kt << 9;            // Kt * 512 elements per 16-col tile
    int nt  = idx / perN;
    int rem = idx - nt * perN;
    int kt  = rem >> 9;
    int r2  = rem & 511;
    int lane = r2 >> 4;
    int j    = r2 & 15;
    int k = kt * 32 + ((lane >> 4) << 3) + (j & 7) + ((j >> 3) << 4);
    int n = nt * 16 + (lane & 15);
    float v = (k < Kx) ? Wx[(size_t)k * 4096 + n]
                       : Wh[(size_t)(k - Kx) * 4096 + n];
    dst[idx] = f2bf(v);
}

__global__ void convert_x_kernel(const float* __restrict__ x,
                                 unsigned short* __restrict__ xbf, int n) {
    int idx = blockIdx.x * blockDim.x + threadIdx.x;
    if (idx < n) xbf[idx] = f2bf(x[idx]);
}

__global__ void init_state_kernel(unsigned short* __restrict__ h,
                                  float* __restrict__ c,
                                  int nh, int nc) {
    int idx = blockIdx.x * blockDim.x + threadIdx.x;
    if (idx < nh) h[idx] = 0;
    if (idx < nc) c[idx] = 0.0f;
}

// ---------------------------------------------------------------------------
// Fused LSTM layer step.
//   grid  = 64 blocks (one per 16-wide slice of H)
//   block = 512 threads = 16 waves; wave w: m-tile = w&3, k-split = w>>2
// Each wave computes ALL FOUR gates for its (m-tile, K-quarter):
//   4 independent accumulator chains; distance-2 software pipeline with
//   TWO NAMED register sets (no runtime-indexed arrays -> no scratch
//   spills), pipeline loop manually unrolled by 2 so ~20 b128 loads are
//   in flight while the XDL pipe runs WMMAs.
// K-split partials reduced through LDS, then fused cell update.
// ---------------------------------------------------------------------------
__global__ __launch_bounds__(512)
void lstm_layer_kernel(const unsigned short* __restrict__ A1, int lda1, int Kt1,
                       const unsigned short* __restrict__ A2, int KtTot,
                       const unsigned short* __restrict__ Bpack,
                       const float* __restrict__ bias,
                       float* __restrict__ c_state,           // [64,1024] this layer
                       unsigned short* __restrict__ h_out_bf,  // [64,1024]
                       float* __restrict__ hseq_out,           // [64,1024] or null
                       float* __restrict__ hfin_out,           // [64,1024] or null
                       float* __restrict__ cfin_out) {         // [64,1024] or null
    __shared__ float part[4][4][64][16];   // [ksplit][gate][batch][col] = 64 KB

    const int tid  = threadIdx.x;
    const int lane = tid & 31;
    const int wid  = tid >> 5;
    const int m    = wid & 3;       // batch tile (wave-uniform)
    const int ks   = __builtin_amdgcn_readfirstlane(wid >> 2);  // K quarter, scalar
    const int h0   = blockIdx.x << 4;

    const int row = m * 16 + (lane & 15);  // batch row 0..63
    const int kbA = (lane >> 4) * 8;       // A-fragment k base within 32-k tile

    const int KtQ = KtTot >> 2;
    const int lo  = ks * KtQ;              // scalar
    const int hi  = lo + KtQ;              // scalar
    const int hi1 = (hi < Kt1) ? hi : Kt1; // scalar
    const int lo2 = (lo > Kt1) ? lo : Kt1; // scalar

    // per-lane base pointers; kt strides are constants -> strength-reduced
    const size_t colStride = (size_t)KtTot * 512;   // bf16 elems per 16-col tile
    const unsigned short* pB[4];
#pragma unroll
    for (int g = 0; g < 4; ++g)
        pB[g] = Bpack + (size_t)(g * 64 + blockIdx.x) * colStride + lane * 16;
    const unsigned short* pA1 = A1 + row * lda1 + kbA;
    const unsigned short* pA2 = A2 + row * 1024 + kbA;

    v8f acc[4];
#pragma unroll
    for (int g = 0; g < 4; ++g)
        acc[g] = (v8f){0.f, 0.f, 0.f, 0.f, 0.f, 0.f, 0.f, 0.f};

    auto loadA = [&](const unsigned short* __restrict__ pa, int ktl, Frag& fa) {
        fa.q[0] = *reinterpret_cast<const uint4*>(pa + ktl * 32);
        fa.q[1] = *reinterpret_cast<const uint4*>(pa + ktl * 32 + 16);
    };
    auto loadB4 = [&](int kt, Frag* fb) {
#pragma unroll
        for (int g = 0; g < 4; ++g) {
            fb[g].q[0] = *reinterpret_cast<const uint4*>(pB[g] + (size_t)kt * 512);
            fb[g].q[1] = *reinterpret_cast<const uint4*>(pB[g] + (size_t)kt * 512 + 8);
        }
    };
    auto wmma4 = [&](const Frag& fa, const Frag* fb) {
#pragma unroll
        for (int g = 0; g < 4; ++g)
            acc[g] = __builtin_amdgcn_wmma_f32_16x16x32_bf16(
                false, fa.v, false, fb[g].v, (short)0, acc[g], false, false);
    };

    // distance-2 software-pipelined GEMM phase over k-tiles [ktFrom, ktTo).
    // Two named fragment sets, pipeline loop unrolled by 2 (no runtime
    // indexing into fragment storage). A index is (kt - aShift), B index kt.
    auto run_phase = [&](const unsigned short* __restrict__ pa,
                         int ktFrom, int ktTo, int aShift) {
        const int n = ktTo - ktFrom;
        if (n <= 0) return;
        Frag fa0, fa1;
        Frag fb0[4], fb1[4];
        loadA(pa, ktFrom - aShift, fa0);
        loadB4(ktFrom, fb0);
        if (n > 1) {
            loadA(pa, ktFrom + 1 - aShift, fa1);
            loadB4(ktFrom + 1, fb1);
        }
        int kt = ktFrom;
        for (; kt + 3 < ktTo; kt += 2) {
            wmma4(fa0, fb0);                     // compute k-tile kt
            loadA(pa, kt + 2 - aShift, fa0);     // refill set0 for kt+2
            loadB4(kt + 2, fb0);                 // (WAR keeps loads after WMMAs)
            wmma4(fa1, fb1);                     // compute k-tile kt+1
            loadA(pa, kt + 3 - aShift, fa1);     // refill set1 for kt+3
            loadB4(kt + 3, fb1);
        }
        const int rem = ktTo - kt;               // 1, 2 or 3
        wmma4(fa0, fb0);
        if (rem == 3) {
            loadA(pa, kt + 2 - aShift, fa0);
            loadB4(kt + 2, fb0);
        }
        if (rem >= 2) wmma4(fa1, fb1);
        if (rem == 3) wmma4(fa0, fb0);
    };

    run_phase(pA1, lo, hi1, 0);     // phase 1: x_t or h of layer below
    run_phase(pA2, lo2, hi, Kt1);   // phase 2: recurrent h, previous step

    // spill partials to LDS (C/D layout: vgpr r -> row r + 8*(lane>>4), col lane&15)
    const int coln  = lane & 15;
    const int rbase = m * 16 + 8 * (lane >> 4);
#pragma unroll
    for (int g = 0; g < 4; ++g)
#pragma unroll
        for (int r = 0; r < 8; ++r)
            part[ks][g][rbase + r][coln] = acc[g][r];

    __syncthreads();

    // K-split reduction + bias + fused cell update: 1024 elems, 2 per thread
#pragma unroll
    for (int it = 0; it < 2; ++it) {
        int e  = tid + it * 512;
        int b  = e >> 4;
        int hh = e & 15;
        float z[4];
#pragma unroll
        for (int g = 0; g < 4; ++g)
            z[g] = part[0][g][b][hh] + part[1][g][b][hh]
                 + part[2][g][b][hh] + part[3][g][b][hh]
                 + bias[g * 1024 + h0 + hh];
        int   ci = b * 1024 + h0 + hh;
        float c_old = c_state[ci];
        float c_new = sigmoidf_fast(z[1]) * c_old
                    + sigmoidf_fast(z[0]) * tanhf_fast(z[2]);
        float h_new = sigmoidf_fast(z[3]) * tanhf_fast(c_new);
        c_state[ci]  = c_new;
        h_out_bf[ci] = f2bf(h_new);
        if (hseq_out) hseq_out[ci] = h_new;
        if (hfin_out) hfin_out[ci] = h_new;
        if (cfin_out) cfin_out[ci] = c_new;
    }
}

// ---------------------------------------------------------------------------
// Host-side orchestration
// ---------------------------------------------------------------------------
extern "C" void kernel_launch(void* const* d_in, const int* in_sizes, int n_in,
                              void* d_out, int out_size, void* d_ws, size_t ws_size,
                              hipStream_t stream) {
    (void)in_sizes; (void)n_in; (void)out_size; (void)ws_size;

    constexpr int S = 512, B = 64, I = 512, H = 1024, L = 4;
    constexpr size_t N4H = 4096;

    const float* x   = (const float*)d_in[0];
    const float* Wx0 = (const float*)d_in[1];
    const float* Wh0 = (const float*)d_in[2];
    const float* b0  = (const float*)d_in[3];
    const float* Wxr = (const float*)d_in[4];
    const float* Whr = (const float*)d_in[5];
    const float* br  = (const float*)d_in[6];
    float* out = (float*)d_out;

    // workspace layout (bytes, 256-aligned sections)
    //   packed weights bf16 : 31,457,280 elts = 62,914,560 B
    //   x bf16              : 16,777,216 elts = 33,554,432 B
    //   h ping-pong bf16    :    524,288 elts =  1,048,576 B
    //   c state f32         :    262,144 elts =  1,048,576 B
    char* base = (char*)d_ws;
    unsigned short* wpack = (unsigned short*)base;
    unsigned short* xbf   = (unsigned short*)(base + 62914560);
    unsigned short* hbuf  = (unsigned short*)(base + 62914560 + 33554432);
    float*          cst   = (float*)(base + 62914560 + 33554432 + 1048576);

    const size_t wOff[4] = {0, 6291456, 14680064, 23068672}; // elements

    // ---- one-time prep ----
    {
        int total0 = 6291456;   // 256 col-tiles * 48 k-tiles * 512
        pack_weights_kernel<<<(total0 + 255) / 256, 256, 0, stream>>>(
            Wx0, Wh0, wpack + wOff[0], 512, 1536, total0);
        int totalr = 8388608;   // 256 * 64 * 512
        for (int l = 1; l < L; ++l) {
            pack_weights_kernel<<<(totalr + 255) / 256, 256, 0, stream>>>(
                Wxr + (size_t)(l - 1) * H * N4H,
                Whr + (size_t)(l - 1) * H * N4H,
                wpack + wOff[l], 1024, 2048, totalr);
        }
        int nx = S * B * I;
        convert_x_kernel<<<(nx + 255) / 256, 256, 0, stream>>>(x, xbf, nx);
        int nh = 2 * L * B * H, nc = L * B * H;
        init_state_kernel<<<(nh + 255) / 256, 256, 0, stream>>>(hbuf, cst, nh, nc);
    }

    // ---- recurrence: S steps x L layers ----
    const size_t SBH = (size_t)S * B * H;
    const size_t LBH = (size_t)L * B * H;
    for (int t = 0; t < S; ++t) {
        const int rp = t & 1;
        const int wp = rp ^ 1;
        for (int l = 0; l < L; ++l) {
            const unsigned short* A1 =
                (l == 0) ? (xbf + (size_t)t * B * I)
                         : (hbuf + ((size_t)(wp * L + (l - 1)) * B) * H);
            const int lda1 = (l == 0) ? I : H;
            const int Kt1  = (l == 0) ? 16 : 32;
            const int KtTot = (l == 0) ? 48 : 64;
            const unsigned short* A2 = hbuf + ((size_t)(rp * L + l) * B) * H;
            const unsigned short* Bp = wpack + wOff[l];
            const float* bias = (l == 0) ? b0 : (br + (size_t)(l - 1) * N4H);
            float* cs = cst + (size_t)l * B * H;
            unsigned short* hout = hbuf + ((size_t)(wp * L + l) * B) * H;
            float* hseq = (l == L - 1) ? (out + (size_t)t * B * H) : nullptr;
            float* hfin = (t == S - 1) ? (out + SBH + (size_t)l * B * H) : nullptr;
            float* cfin = (t == S - 1) ? (out + SBH + LBH + (size_t)l * B * H) : nullptr;

            lstm_layer_kernel<<<dim3(64), dim3(512), 0, stream>>>(
                A1, lda1, Kt1, A2, KtTot, Bp, bias, cs, hout, hseq, hfin, cfin);
        }
    }
}
